// Attention_12197707120711
// MI455X (gfx1250) — matrix operand order repaired and checked by reading, not compiled, not run on hardware
//
#include <hip/hip_runtime.h>
#include <hip/hip_bf16.h>

// ---------------------------------------------------------------------------
// MHA block, B=8 P=1024 D=512 H=8 DH=64, bf16 WMMA (V_WMMA_F32_16X16X32_BF16)
// Softmax over the QUERY axis (axis=-2): column softmax of S[i,j].
//   out = A @ (V .* (1/Z)_rows),  A[i,j] = exp(S[i,j]-m[j]),
//   m[j] = max_i S[i,j], Z[j] = sum_i exp(S[i,j]-m[j]).
// Data movement: f32->bf16 via float4 loads + packed converts; bf16 tiles
// staged with GLOBAL_LOAD_ASYNC_TO_LDS_B128 (ASYNCcnt) where contiguous.
// ---------------------------------------------------------------------------

typedef __attribute__((ext_vector_type(16))) __bf16 v16bf;
typedef __attribute__((ext_vector_type(8)))  __bf16 v8bf;
typedef __attribute__((ext_vector_type(4)))  __bf16 v4bf;
typedef __attribute__((ext_vector_type(8)))  float  v8f;

__device__ __forceinline__ v8f wmma_bf16f32(v16bf a, v16bf b, v8f c) {
  return __builtin_amdgcn_wmma_f32_16x16x32_bf16(false, a, false, b,
                                                 (short)0, c, false, false);
}

union Frag16 { v16bf v; v8bf h[2]; };

// A fragment (16x32 bf16), LDS row-major [M][LDK].
__device__ __forceinline__ v16bf load_a_frag(const __bf16* lds, int ldk,
                                             int m0, int kk, int lane) {
  const int m  = m0 + (lane & 15);
  const int kb = kk + ((lane >> 4) << 3);
  const __bf16* p = lds + m * ldk + kb;
  Frag16 f;
  f.h[0] = *(const v8bf*)(p);
  f.h[1] = *(const v8bf*)(p + 16);
  return f.v;
}

// B fragment (32x16 bf16), LDS stored N-major [N][LDK] (Bt[n][k]).
__device__ __forceinline__ v16bf load_b_frag(const __bf16* ldsBt, int ldk,
                                             int n0, int kk, int lane) {
  const int n  = n0 + (lane & 15);
  const int kb = kk + ((lane >> 4) << 3);
  const __bf16* p = ldsBt + n * ldk + kb;
  Frag16 f;
  f.h[0] = *(const v8bf*)(p);
  f.h[1] = *(const v8bf*)(p + 16);
  return f.v;
}

// Async global->LDS 128-bit copy (CDNA5, tracked with ASYNCcnt).
// Generic LDS pointer: low 32 bits are the LDS byte offset.
__device__ __forceinline__ void async_g2l_b128(const void* gptr, void* lptr) {
  unsigned lds = (unsigned)(unsigned long long)lptr;
  unsigned long long ga = (unsigned long long)gptr;
  asm volatile("global_load_async_to_lds_b128 %0, %1, off"
               :: "v"(lds), "v"(ga) : "memory");
}
__device__ __forceinline__ void wait_async0() {
  asm volatile("s_wait_asynccnt 0" ::: "memory");
}

__device__ __forceinline__ v4bf cvt4(float a, float b, float c, float d) {
  v4bf o; o[0] = (__bf16)a; o[1] = (__bf16)b; o[2] = (__bf16)c; o[3] = (__bf16)d;
  return o;
}

// ---------------------------------------------------------------------------
// Kernel 1: QKV projection. X[8192][512] f32, W[512][1536] f32.
// Output scattered as bf16 in layout [which(3)][B][H][P][DH].
// Macro tile 64(M) x 256(N), waves 2x4, 32x64 per wave (8 WMMA / k-step).
// ---------------------------------------------------------------------------
__global__ __launch_bounds__(256) void k_qkv_gemm(const float* __restrict__ X,
                                                  const float* __restrict__ W,
                                                  __bf16* __restrict__ QKV) {
  __shared__ __attribute__((aligned(32))) __bf16 ldsA[64 * 32];    //  4 KiB
  __shared__ __attribute__((aligned(32))) __bf16 ldsBt[256 * 32];  // 16 KiB
  const int tid = threadIdx.x, lane = tid & 31, w = tid >> 5;
  const int m0 = (w >> 2) * 32, n0 = (w & 3) * 64;
  const int rowBase = blockIdx.y * 64;
  const int colBase = blockIdx.x * 256;
  v8f acc[2][4] = {};

  for (int kt = 0; kt < 512; kt += 32) {
    // A: 64x32 f32 -> bf16, float4 loads (2 per thread)
    for (int e = tid; e < 512; e += 256) {
      int r = e >> 3, k4 = (e & 7) * 4;
      float4 f = *(const float4*)(X + (size_t)(rowBase + r) * 512 + kt + k4);
      *(v4bf*)(ldsA + e * 4) = cvt4(f.x, f.y, f.z, f.w);
    }
    // Bt: 256 cols x 32 k, transposed store; float4 loads along n (8/thread)
    for (int e = tid; e < 2048; e += 256) {
      int n4 = (e & 63) * 4, k = e >> 6;
      float4 f = *(const float4*)(W + (size_t)(kt + k) * 1536 + colBase + n4);
      ldsBt[(n4 + 0) * 32 + k] = (__bf16)f.x;
      ldsBt[(n4 + 1) * 32 + k] = (__bf16)f.y;
      ldsBt[(n4 + 2) * 32 + k] = (__bf16)f.z;
      ldsBt[(n4 + 3) * 32 + k] = (__bf16)f.w;
    }
    __syncthreads();
    v16bf a0 = load_a_frag(ldsA, 32, m0, 0, lane);
    v16bf a1 = load_a_frag(ldsA, 32, m0 + 16, 0, lane);
    for (int ni = 0; ni < 4; ++ni) {
      v16bf b = load_b_frag(ldsBt, 32, n0 + ni * 16, 0, lane);
      acc[0][ni] = wmma_bf16f32(a0, b, acc[0][ni]);
      acc[1][ni] = wmma_bf16f32(a1, b, acc[1][ni]);
    }
    __syncthreads();
  }
  for (int mi = 0; mi < 2; ++mi)
    for (int ni = 0; ni < 4; ++ni)
      for (int r = 0; r < 8; ++r) {
        int row = rowBase + m0 + mi * 16 + r + ((lane >> 4) << 3);
        int col = colBase + n0 + ni * 16 + (lane & 15);
        int b = row >> 10, p = row & 1023;
        int which = col >> 9, rem = col & 511, h = rem >> 6, d = rem & 63;
        size_t addr = (((size_t)which * 8 + b) * 8 + h) * 65536 +
                      (size_t)p * 64 + d;
        QKV[addr] = (__bf16)acc[mi][ni][r];
      }
}

// ---------------------------------------------------------------------------
// Kernel 2: S = Q @ K^T * scale for one batch element.
// Qb,Kb: [H][1024][64] bf16 (contiguous tiles -> async B128 to LDS).
// S: [H][1024][1024] f32.  Macro tile 64(i) x 128(j), K-depth 64.
// ---------------------------------------------------------------------------
__global__ __launch_bounds__(256) void k_qk_gemm(const __bf16* __restrict__ Qb,
                                                 const __bf16* __restrict__ Kb,
                                                 float* __restrict__ S) {
  __shared__ __attribute__((aligned(32))) __bf16 ldsA[64 * 64];    //  8 KiB
  __shared__ __attribute__((aligned(32))) __bf16 ldsBt[128 * 64];  // 16 KiB
  const int tid = threadIdx.x, lane = tid & 31, w = tid >> 5;
  const int m0 = (w >> 2) * 32, n0 = (w & 3) * 32;
  const int h = blockIdx.z;
  const int iBase = blockIdx.y * 64;
  const int jBase = blockIdx.x * 128;

  // Async DMA-style staging: contiguous bf16 regions, 16B chunks.
  const __bf16* srcA = Qb + (size_t)h * 65536 + (size_t)iBase * 64;
  for (int c = tid; c < 512; c += 256)          // 8 KiB = 512 x 16B
    async_g2l_b128(srcA + c * 8, ldsA + c * 8);
  const __bf16* srcB = Kb + (size_t)h * 65536 + (size_t)jBase * 64;
  for (int c = tid; c < 1024; c += 256)         // 16 KiB = 1024 x 16B
    async_g2l_b128(srcB + c * 8, ldsBt + c * 8);
  wait_async0();
  __syncthreads();

  v8f acc[2][2] = {};
  for (int kk = 0; kk < 64; kk += 32) {
    v16bf a0 = load_a_frag(ldsA, 64, m0, kk, lane);
    v16bf a1 = load_a_frag(ldsA, 64, m0 + 16, kk, lane);
    v16bf b0 = load_b_frag(ldsBt, 64, n0, kk, lane);
    v16bf b1 = load_b_frag(ldsBt, 64, n0 + 16, kk, lane);
    acc[0][0] = wmma_bf16f32(a0, b0, acc[0][0]);
    acc[0][1] = wmma_bf16f32(a0, b1, acc[0][1]);
    acc[1][0] = wmma_bf16f32(a1, b0, acc[1][0]);
    acc[1][1] = wmma_bf16f32(a1, b1, acc[1][1]);
  }
  const float scale = 0.125f;  // DH^-0.5
  for (int mi = 0; mi < 2; ++mi)
    for (int ni = 0; ni < 2; ++ni)
      for (int r = 0; r < 8; ++r) {
        int i = iBase + m0 + mi * 16 + r + ((lane >> 4) << 3);
        int j = jBase + n0 + ni * 16 + (lane & 15);
        S[(size_t)h * 1048576 + (size_t)i * 1024 + j] = acc[mi][ni][r] * scale;
      }
}

// ---------------------------------------------------------------------------
// Kernel 3: column stats over the query axis. gid = h*1024 + j.
// ---------------------------------------------------------------------------
__global__ __launch_bounds__(256) void k_col_stats(const float* __restrict__ S,
                                                   float* __restrict__ mOut,
                                                   float* __restrict__ rZOut) {
  int gid = blockIdx.x * 256 + threadIdx.x;
  int h = gid >> 10, j = gid & 1023;
  const float* col = S + (size_t)h * 1048576 + j;
  float m = -3.0e38f;
  for (int i = 0; i < 1024; ++i) m = fmaxf(m, col[(size_t)i * 1024]);
  float zsum = 0.0f;
  for (int i = 0; i < 1024; ++i) zsum += __expf(col[(size_t)i * 1024] - m);
  mOut[gid] = m;
  rZOut[gid] = 1.0f / zsum;
}

// ---------------------------------------------------------------------------
// Kernel 4: out = exp(S - m_cols) @ (V .* rZ_rows) for one batch element.
// Macro tile 128(i) x 64(d), waves 4x2. attn: [B][P][H*DH] bf16.
// ---------------------------------------------------------------------------
__global__ __launch_bounds__(256) void k_av_gemm(const float* __restrict__ S,
                                                 const __bf16* __restrict__ Vb,
                                                 const float* __restrict__ mArr,
                                                 const float* __restrict__ rZArr,
                                                 __bf16* __restrict__ attn,
                                                 int bIdx) {
  __shared__ __attribute__((aligned(32))) __bf16 ldsA[128 * 32];  // 8 KiB
  __shared__ __attribute__((aligned(32))) __bf16 ldsBt[64 * 32];  // 4 KiB
  const int tid = threadIdx.x, lane = tid & 31, w = tid >> 5;
  const int m0 = (w >> 1) * 32, n0 = (w & 1) * 32;
  const int h = blockIdx.z;
  const int iBase = blockIdx.y * 128;
  v8f acc[2][2] = {};

  for (int j0 = 0; j0 < 1024; j0 += 32) {
    // A: exp(S - m) -> bf16, float4 loads (4 per thread)
    for (int e = tid; e < 1024; e += 256) {
      int i = e >> 3, j4 = (e & 7) * 4;
      float4 s =
          *(const float4*)(S + (size_t)h * 1048576 +
                           (size_t)(iBase + i) * 1024 + j0 + j4);
      float4 mm = *(const float4*)(mArr + h * 1024 + j0 + j4);
      *(v4bf*)(ldsA + e * 4) = cvt4(__expf(s.x - mm.x), __expf(s.y - mm.y),
                                    __expf(s.z - mm.z), __expf(s.w - mm.w));
    }
    // Bt[d][j] = V[j][d] * rZ[j], bf16x4 loads along d (2 per thread)
    for (int e = tid; e < 512; e += 256) {
      int d4 = (e & 15) * 4, j = e >> 4;
      v4bf v = *(const v4bf*)(Vb + (size_t)h * 65536 +
                              (size_t)(j0 + j) * 64 + d4);
      float rz = rZArr[h * 1024 + j0 + j];
      ldsBt[(d4 + 0) * 32 + j] = (__bf16)((float)v[0] * rz);
      ldsBt[(d4 + 1) * 32 + j] = (__bf16)((float)v[1] * rz);
      ldsBt[(d4 + 2) * 32 + j] = (__bf16)((float)v[2] * rz);
      ldsBt[(d4 + 3) * 32 + j] = (__bf16)((float)v[3] * rz);
    }
    __syncthreads();
    v16bf a0 = load_a_frag(ldsA, 32, m0, 0, lane);
    v16bf a1 = load_a_frag(ldsA, 32, m0 + 16, 0, lane);
    v16bf b0 = load_b_frag(ldsBt, 32, n0, 0, lane);
    v16bf b1 = load_b_frag(ldsBt, 32, n0 + 16, 0, lane);
    acc[0][0] = wmma_bf16f32(a0, b0, acc[0][0]);
    acc[0][1] = wmma_bf16f32(a0, b1, acc[0][1]);
    acc[1][0] = wmma_bf16f32(a1, b0, acc[1][0]);
    acc[1][1] = wmma_bf16f32(a1, b1, acc[1][1]);
    __syncthreads();
  }
  for (int mi = 0; mi < 2; ++mi)
    for (int ni = 0; ni < 2; ++ni)
      for (int r = 0; r < 8; ++r) {
        int p = iBase + m0 + mi * 16 + r + ((lane >> 4) << 3);
        int d = n0 + ni * 16 + (lane & 15);
        attn[(size_t)bIdx * 524288 + (size_t)p * 512 + h * 64 + d] =
            (__bf16)acc[mi][ni][r];
      }
}

// ---------------------------------------------------------------------------
// Kernel 5: final projection Y = attn @ W_out + b_out.
// attn[8192][512] bf16 (async B128 staging), W[512][512] f32, Y f32.
// Macro tile 64(M) x 256(N), waves 2x4, 32x64 per wave.
// ---------------------------------------------------------------------------
__global__ __launch_bounds__(256) void k_out_gemm(const __bf16* __restrict__ A,
                                                  const float* __restrict__ W,
                                                  const float* __restrict__ bias,
                                                  float* __restrict__ Y) {
  __shared__ __attribute__((aligned(32))) __bf16 ldsA[64 * 32];    //  4 KiB
  __shared__ __attribute__((aligned(32))) __bf16 ldsBt[256 * 32];  // 16 KiB
  const int tid = threadIdx.x, lane = tid & 31, w = tid >> 5;
  const int m0 = (w >> 2) * 32, n0 = (w & 3) * 64;
  const int rowBase = blockIdx.y * 64;
  const int colBase = blockIdx.x * 256;
  v8f acc[2][4] = {};

  for (int kt = 0; kt < 512; kt += 32) {
    // A: 64 rows x 32 bf16 = 64B/row -> 4 async B128 chunks per row
    if (tid < 256) {
      int r = tid >> 2, cc = (tid & 3) * 8;
      async_g2l_b128(A + (size_t)(rowBase + r) * 512 + kt + cc,
                     ldsA + r * 32 + cc);
    }
    // Bt: f32 W transposed+converted; float4 loads along n (8/thread)
    for (int e = tid; e < 2048; e += 256) {
      int n4 = (e & 63) * 4, k = e >> 6;
      float4 f = *(const float4*)(W + (size_t)(kt + k) * 512 + colBase + n4);
      ldsBt[(n4 + 0) * 32 + k] = (__bf16)f.x;
      ldsBt[(n4 + 1) * 32 + k] = (__bf16)f.y;
      ldsBt[(n4 + 2) * 32 + k] = (__bf16)f.z;
      ldsBt[(n4 + 3) * 32 + k] = (__bf16)f.w;
    }
    wait_async0();
    __syncthreads();
    v16bf a0 = load_a_frag(ldsA, 32, m0, 0, lane);
    v16bf a1 = load_a_frag(ldsA, 32, m0 + 16, 0, lane);
    for (int ni = 0; ni < 4; ++ni) {
      v16bf b = load_b_frag(ldsBt, 32, n0 + ni * 16, 0, lane);
      acc[0][ni] = wmma_bf16f32(a0, b, acc[0][ni]);
      acc[1][ni] = wmma_bf16f32(a1, b, acc[1][ni]);
    }
    __syncthreads();
  }
  for (int mi = 0; mi < 2; ++mi)
    for (int ni = 0; ni < 4; ++ni)
      for (int r = 0; r < 8; ++r) {
        int row = rowBase + m0 + mi * 16 + r + ((lane >> 4) << 3);
        int col = colBase + n0 + ni * 16 + (lane & 15);
        Y[(size_t)row * 512 + col] = acc[mi][ni][r] + bias[col];
      }
}

// ---------------------------------------------------------------------------
// Host-side orchestration. Workspace layout (bytes):
//   [0, 24M)   qkv bf16 [3][8][8][1024][64]
//   [24M,32M)  attn bf16 [8][1024][512]
//   [32M,64M)  S f32 [8][1024][1024] (per-batch chunk, reused)
//   [64M,..)   m f32 [8192], rZ f32 [8192]
// ---------------------------------------------------------------------------
extern "C" void kernel_launch(void* const* d_in, const int* in_sizes, int n_in,
                              void* d_out, int out_size, void* d_ws,
                              size_t ws_size, hipStream_t stream) {
  const float* x    = (const float*)d_in[0];
  const float* Wqkv = (const float*)d_in[1];
  const float* Wout = (const float*)d_in[2];
  const float* bout = (const float*)d_in[3];
  float* out = (float*)d_out;

  char* ws = (char*)d_ws;
  const size_t QKV_BYTES  = (size_t)3 * 8 * 8 * 1024 * 64 * 2;  // 24 MiB
  const size_t ATTN_BYTES = (size_t)8 * 1024 * 512 * 2;         //  8 MiB
  const size_t S_BYTES    = (size_t)8 * 1024 * 1024 * 4;        // 32 MiB
  __bf16* qkv  = (__bf16*)ws;
  __bf16* attn = (__bf16*)(ws + QKV_BYTES);
  float*  S    = (float*)(ws + QKV_BYTES + ATTN_BYTES);
  float*  mArr = (float*)(ws + QKV_BYTES + ATTN_BYTES + S_BYTES);
  float*  rZ   = mArr + 8 * 1024;

  // 1) QKV projection: grid = (1536/256) x (8192/64)
  k_qkv_gemm<<<dim3(6, 128), 256, 0, stream>>>(x, Wqkv, qkv);

  // 2-4) attention, chunked per batch element (stream order serializes reuse)
  const size_t perWhich = (size_t)8 * 8 * 65536;
  for (int b = 0; b < 8; ++b) {
    const __bf16* Qb = qkv + 0 * perWhich + (size_t)b * 8 * 65536;
    const __bf16* Kb = qkv + 1 * perWhich + (size_t)b * 8 * 65536;
    const __bf16* Vb = qkv + 2 * perWhich + (size_t)b * 8 * 65536;
    k_qk_gemm<<<dim3(8, 16, 8), 256, 0, stream>>>(Qb, Kb, S);
    k_col_stats<<<dim3(32), 256, 0, stream>>>(S, mArr, rZ);
    k_av_gemm<<<dim3(1, 8, 8), 256, 0, stream>>>(S, Vb, mArr, rZ, attn, b);
  }

  // 5) output projection: grid = (512/256) x (8192/64)
  k_out_gemm<<<dim3(2, 128), 256, 0, stream>>>(attn, Wout, bout, out);
}
